// Mutation_MIL_MoransI_60275571032651
// MI455X (gfx1250) — matrix operand order repaired and checked by reading, not compiled, not run on hardware
//
#include <hip/hip_runtime.h>

// ---------------------------------------------------------------------------
// Mutation_MIL_MoransI for MI455X (gfx1250, wave32, WMMA)
//
// Dominant cost: h = tanh(x[50000,2048] @ W1[2048,128] + b1).
// 26.2 GFLOP vs 410 MB mandatory fp32 read of x  =>  ~64 FLOP/byte.
// bf16 WMMA (16x16x32, f32 accum) keeps this memory-bound at 23.3 TB/s.
// W1 is pre-transposed once into a bf16 [d][k] image (512 KB, L2-resident),
// so BOTH operands stream global->LDS via the gfx1250 async-LDS path
// (ASYNCcnt) into double-buffered LDS images with no VGPR transit.
// fp32->bf16 conversion of x happens at A-fragment build (co-executes with
// the XDL WMMA chain).
// ---------------------------------------------------------------------------

typedef __bf16 bf16_t;
typedef bf16_t v16bf __attribute__((ext_vector_type(16)));
typedef float  v8f   __attribute__((ext_vector_type(8)));
typedef int    v4i_t __attribute__((ext_vector_type(4)));

#define N_TILE 50000
#define GXY    256
#define CELLS  (GXY * GXY)
#define L      2048
#define D      128
#define NOUT   7

#define ROWS_PER_WG 128   // 8 waves x 16 rows
#define KC          32    // K per WMMA chunk
#define NKC         (L / KC)

#if __has_builtin(__builtin_amdgcn_global_load_async_to_lds_b128) && \
    __has_builtin(__builtin_amdgcn_s_wait_asynccnt)
#define HAVE_ASYNC_LDS 1
typedef __attribute__((address_space(1))) v4i_t* g_v4i_p;  // global int4*
typedef __attribute__((address_space(3))) v4i_t* l_v4i_p;  // LDS int4*
#define ASYNC_COPY16(g, l)                                                   \
  __builtin_amdgcn_global_load_async_to_lds_b128(                            \
      (g_v4i_p)(v4i_t*)(g), (l_v4i_p)(v4i_t*)(l), 0, 0)
#define ASYNC_WAIT() __builtin_amdgcn_s_wait_asynccnt(0)
#else
#define HAVE_ASYNC_LDS 0
#define ASYNC_WAIT()
#endif

// ---------------------------------------------------------------------------
// K0: one-shot W1 transpose+convert: w1tg[d*L + k] = bf16(W1[k*D + d])
// ---------------------------------------------------------------------------
__global__ void w1_transpose_kernel(const float* __restrict__ W1,
                                    bf16_t* __restrict__ w1tg)
{
  int i = blockIdx.x * blockDim.x + threadIdx.x;   // i = d*L + k
  if (i >= L * D) return;
  int d = i >> 11, k = i & (L - 1);
  w1tg[i] = (bf16_t)W1[(size_t)k * D + d];
}

// ---------------------------------------------------------------------------
// K1: fused  s_pre[n] = tanh( tanh(x[n,:]@W1 + b1) @ W2 + b2 )
// ---------------------------------------------------------------------------
__global__ __launch_bounds__(256)
void attn_gemm_kernel(const float* __restrict__ x,
                      const bf16_t* __restrict__ w1tg,   // bf16 W1^T, [d][k]
                      const float* __restrict__ b1,
                      const float* __restrict__ W2,
                      const float* __restrict__ b2,
                      float* __restrict__ s_pre)
{
  __shared__ alignas(16) float  xsf[2][ROWS_PER_WG * KC];  // fp32 x image, 2x16KB
  __shared__ alignas(16) bf16_t w1t[2][D * KC];            // bf16 W1^T image, 2x8KB
  __shared__ float hbuf[ROWS_PER_WG * D];                  // tanh(h), 64KB

  const int tid   = threadIdx.x;
  const int wave  = tid >> 5;
  const int lane  = tid & 31;
  const int hsel  = lane >> 4;   // half-wave select (ISA 7.12.2 layouts)
  const int lan16 = lane & 15;
  const int n0    = blockIdx.x * ROWS_PER_WG;

  // per-thread staging coordinates
  // x  : idx4  = tid + s*256 (s<4) -> row = idx4>>3,  kq = (idx4&7)*4   (16B)
  // w1 : idx16 = tid + s*256 (s<2) -> d   = idx16>>2, kg = (idx16&3)*8  (16B)

  v8f acc[8] = {};   // 16 rows x 128 cols per wave (8 N-tiles of 16)

  // ---- prologue: stage chunk 0 into buffer 0 -------------------------------
  {
#if HAVE_ASYNC_LDS
    #pragma unroll
    for (int s = 0; s < 4; ++s) {
      int idx4 = tid + s * 256, r = idx4 >> 3, kq = (idx4 & 7) * 4;
      int n = n0 + r; n = n < N_TILE ? n : N_TILE - 1;   // clamp tail rows
      ASYNC_COPY16(&x[(size_t)n * L + kq], &xsf[0][r * KC + kq]);
    }
    #pragma unroll
    for (int s = 0; s < 2; ++s) {
      int idx16 = tid + s * 256, d = idx16 >> 2, kg = (idx16 & 3) * 8;
      ASYNC_COPY16(&w1tg[(size_t)d * L + kg], &w1t[0][d * KC + kg]);
    }
#else
    #pragma unroll
    for (int s = 0; s < 4; ++s) {
      int idx4 = tid + s * 256, r = idx4 >> 3, kq = (idx4 & 7) * 4;
      int n = n0 + r; n = n < N_TILE ? n : N_TILE - 1;
      *(float4*)&xsf[0][r * KC + kq] = *(const float4*)&x[(size_t)n * L + kq];
    }
    #pragma unroll
    for (int s = 0; s < 2; ++s) {
      int idx16 = tid + s * 256, d = idx16 >> 2, kg = (idx16 & 3) * 8;
      *(uint4*)&w1t[0][d * KC + kg] = *(const uint4*)&w1tg[(size_t)d * L + kg];
    }
#endif
    ASYNC_WAIT();
    __syncthreads();
  }

  // ---- main K loop: double-buffered, one barrier per chunk -----------------
  for (int kc = 0; kc < NKC; ++kc) {
    const int p = kc & 1, q = p ^ 1;
    const int kn = kc + 1;
#if !HAVE_ASYNC_LDS
    float4 xreg[4];
    uint4  wreg[2];
#endif
    if (kn < NKC) {
      const int kb = kn * KC;
#if HAVE_ASYNC_LDS
      #pragma unroll
      for (int s = 0; s < 4; ++s) {
        int idx4 = tid + s * 256, r = idx4 >> 3, kq = (idx4 & 7) * 4;
        int n = n0 + r; n = n < N_TILE ? n : N_TILE - 1;
        ASYNC_COPY16(&x[(size_t)n * L + kb + kq], &xsf[q][r * KC + kq]);
      }
      #pragma unroll
      for (int s = 0; s < 2; ++s) {
        int idx16 = tid + s * 256, d = idx16 >> 2, kg = (idx16 & 3) * 8;
        ASYNC_COPY16(&w1tg[(size_t)d * L + kb + kg], &w1t[q][d * KC + kg]);
      }
#else
      #pragma unroll
      for (int s = 0; s < 4; ++s) {
        int idx4 = tid + s * 256, r = idx4 >> 3, kq = (idx4 & 7) * 4;
        int n = n0 + r; n = n < N_TILE ? n : N_TILE - 1;
        xreg[s] = *(const float4*)&x[(size_t)n * L + kb + kq];
      }
      #pragma unroll
      for (int s = 0; s < 2; ++s) {
        int idx16 = tid + s * 256, d = idx16 >> 2, kg = (idx16 & 3) * 8;
        wreg[s] = *(const uint4*)&w1tg[(size_t)d * L + kb + kg];
      }
#endif
      if (kn + 1 < NKC) {   // light prefetch two chunks ahead
        int r = tid >> 3, kq = (tid & 7) * 4;
        int n = n0 + r; n = n < N_TILE ? n : N_TILE - 1;
        __builtin_prefetch(&x[(size_t)n * L + (kn + 1) * KC + kq], 0, 1);
      }
    }

    // A fragment (16x32 bf16): lane holds row M=lan16;
    // hsel=0 -> K 0..7 & 16..23, hsel=1 -> K 8..15 & 24..31 (ISA 7.12.2).
    // Convert fp32->bf16 here (co-executes with XDL WMMAs).
    union { v16bf v; bf16_t e[16]; } af;
    {
      const float4* xr = (const float4*)&xsf[p][(wave * 16 + lan16) * KC];
      float4 a0 = xr[hsel * 2], a1 = xr[hsel * 2 + 1];
      float4 a2 = xr[4 + hsel * 2], a3 = xr[5 + hsel * 2];
      af.e[0]  = (bf16_t)a0.x; af.e[1]  = (bf16_t)a0.y;
      af.e[2]  = (bf16_t)a0.z; af.e[3]  = (bf16_t)a0.w;
      af.e[4]  = (bf16_t)a1.x; af.e[5]  = (bf16_t)a1.y;
      af.e[6]  = (bf16_t)a1.z; af.e[7]  = (bf16_t)a1.w;
      af.e[8]  = (bf16_t)a2.x; af.e[9]  = (bf16_t)a2.y;
      af.e[10] = (bf16_t)a2.z; af.e[11] = (bf16_t)a2.w;
      af.e[12] = (bf16_t)a3.x; af.e[13] = (bf16_t)a3.y;
      af.e[14] = (bf16_t)a3.z; af.e[15] = (bf16_t)a3.w;
    }

    // 8 N-tiles: B fragment (32x16 bf16): lane holds col N=t*16+lan16,
    // hsel=0 -> K 0..7 & 16..23, hsel=1 -> K 8..15 & 24..31 (matches A).
    #pragma unroll
    for (int t = 0; t < 8; ++t) {
      union { uint4 qd[2]; v16bf v; } bfr;
      const uint4* wrow = (const uint4*)&w1t[p][(t * 16 + lan16) * KC];
      bfr.qd[0] = wrow[hsel];
      bfr.qd[1] = wrow[2 + hsel];
      acc[t] = __builtin_amdgcn_wmma_f32_16x16x32_bf16(
          false, af.v, false, bfr.v, (short)0, acc[t], false, false);
    }

    if (kn < NKC) {
#if !HAVE_ASYNC_LDS
      #pragma unroll
      for (int s = 0; s < 4; ++s) {
        int idx4 = tid + s * 256, r = idx4 >> 3, kq = (idx4 & 7) * 4;
        *(float4*)&xsf[q][r * KC + kq] = xreg[s];
      }
      #pragma unroll
      for (int s = 0; s < 2; ++s) {
        int idx16 = tid + s * 256, d = idx16 >> 2, kg = (idx16 & 3) * 8;
        *(uint4*)&w1t[q][d * KC + kg] = wreg[s];
      }
#endif
      ASYNC_WAIT();
    }
    __syncthreads();
  }

  // bias + tanh; spill per-row h to LDS.
  // C/D layout: lane half -> M base {0,8}, VGPR j -> M offset, N = lan16.
  #pragma unroll
  for (int t = 0; t < 8; ++t) {
    int col = t * 16 + lan16;
    float bias = b1[col];
    #pragma unroll
    for (int j = 0; j < 8; ++j) {
      int row = wave * 16 + hsel * 8 + j;
      hbuf[row * D + col] = tanhf(acc[t][j] + bias);
    }
  }
  __syncthreads();

  // second layer: per-row dot over D=128 with W2, then tanh
  if (tid < ROWS_PER_WG) {
    int n = n0 + tid;
    float s = b2[0];
    #pragma unroll 8
    for (int d0 = 0; d0 < D; ++d0) s += hbuf[tid * D + d0] * W2[d0];
    if (n < N_TILE) s_pre[n] = tanhf(s);
  }
}

// ---------------------------------------------------------------------------
// K2: softmax stats (max, sum of exp) over N_TILE — single WG
// ---------------------------------------------------------------------------
__global__ __launch_bounds__(1024)
void softmax_stats_kernel(const float* __restrict__ s_pre, float* __restrict__ stats)
{
  __shared__ float red[1024];
  int tid = threadIdx.x;
  float m = -3.402823466e+38f;
  for (int i = tid; i < N_TILE; i += 1024) m = fmaxf(m, s_pre[i]);
  red[tid] = m; __syncthreads();
  for (int s = 512; s > 0; s >>= 1) {
    if (tid < s) red[tid] = fmaxf(red[tid], red[tid + s]);
    __syncthreads();
  }
  float gmax = red[0];
  __syncthreads();
  float sum = 0.f;
  for (int i = tid; i < N_TILE; i += 1024) sum += __expf(s_pre[i] - gmax);
  red[tid] = sum; __syncthreads();
  for (int s = 512; s > 0; s >>= 1) {
    if (tid < s) red[tid] += red[tid + s];
    __syncthreads();
  }
  if (tid == 0) { stats[0] = gmax; stats[1] = red[0]; }
}

// ---------------------------------------------------------------------------
// K3a/K3b: zero A_matrix, then scatter softmax values at unique coords
// ---------------------------------------------------------------------------
__global__ void zero_kernel(float* __restrict__ p, int n)
{
  int i = blockIdx.x * blockDim.x + threadIdx.x;
  if (i < n) p[i] = 0.f;
}

__global__ void scatter_kernel(const float* __restrict__ s_pre,
                               const int* __restrict__ coor,
                               const float* __restrict__ stats,
                               float* __restrict__ Amat)
{
  int i = blockIdx.x * blockDim.x + threadIdx.x;
  if (i >= N_TILE) return;
  float v = __expf(s_pre[i] - stats[0]) / stats[1];
  Amat[coor[2 * i] * GXY + coor[2 * i + 1]] = v;
}

// ---------------------------------------------------------------------------
// K4: sum / min / max over A_matrix — single WG
// ---------------------------------------------------------------------------
__global__ __launch_bounds__(1024)
void mat_stats_kernel(const float* __restrict__ Amat, float* __restrict__ stats)
{
  __shared__ float rs[1024], rmn[1024], rmx[1024];
  int tid = threadIdx.x;
  float s = 0.f, mn = 3.402823466e+38f, mx = -3.402823466e+38f;
  for (int i = tid; i < CELLS; i += 1024) {
    float v = Amat[i];
    s += v; mn = fminf(mn, v); mx = fmaxf(mx, v);
  }
  rs[tid] = s; rmn[tid] = mn; rmx[tid] = mx; __syncthreads();
  for (int st = 512; st > 0; st >>= 1) {
    if (tid < st) {
      rs[tid] += rs[tid + st];
      rmn[tid] = fminf(rmn[tid], rmn[tid + st]);
      rmx[tid] = fmaxf(rmx[tid], rmx[tid + st]);
    }
    __syncthreads();
  }
  if (tid == 0) { stats[2] = rs[0]; stats[3] = rmn[0]; stats[4] = rmx[0]; }
}

// ---------------------------------------------------------------------------
// K5: Moran's I sums (rook contiguity, row-standardized) + bucket counts
// ---------------------------------------------------------------------------
__global__ __launch_bounds__(1024)
void moran_bucket_kernel(const float* __restrict__ Amat, float* __restrict__ stats)
{
  __shared__ float rnum[1024], rden[1024];
  __shared__ int cnt[4];
  int tid = threadIdx.x;
  if (tid < 4) cnt[tid] = 0;
  __syncthreads();

  const float mean = stats[2] * (1.0f / (float)CELLS);
  const float fmn = stats[3];
  const float fmx6 = stats[4] + 1e-6f;
  const float t1 = fmn + (fmx6 - fmn) * (1.f / 3.f);
  const float t2 = fmn + (fmx6 - fmn) * (2.f / 3.f);

  float num = 0.f, den = 0.f;
  int c0 = 0, c1 = 0, c2 = 0, c3 = 0;
  for (int i = tid; i < CELLS; i += 1024) {
    float v = Amat[i];
    int gx = i >> 8, gy = i & 255;
    float z = v - mean;
    float ns = 0.f; int nc = 0;
    if (gx > 0)       { ns += Amat[i - GXY] - mean; ++nc; }
    if (gx < GXY - 1) { ns += Amat[i + GXY] - mean; ++nc; }
    if (gy > 0)       { ns += Amat[i - 1]   - mean; ++nc; }
    if (gy < GXY - 1) { ns += Amat[i + 1]   - mean; ++nc; }
    float lag = ns / (float)nc;
    num += z * lag;
    den += z * z;
    // searchsorted(right) over [fmn, t1, t2, fmx6] -> values in {1,2,3}
    int b = (v >= fmn) + (v >= t1) + (v >= t2) + (v >= fmx6);
    b = b > 3 ? 3 : b;
    c0 += (b == 0); c1 += (b == 1); c2 += (b == 2); c3 += (b == 3);
  }
  rnum[tid] = num; rden[tid] = den; __syncthreads();
  for (int st = 512; st > 0; st >>= 1) {
    if (tid < st) { rnum[tid] += rnum[tid + st]; rden[tid] += rden[tid + st]; }
    __syncthreads();
  }
  if (c0) atomicAdd(&cnt[0], c0);
  if (c1) atomicAdd(&cnt[1], c1);
  if (c2) atomicAdd(&cnt[2], c2);
  if (c3) atomicAdd(&cnt[3], c3);
  __syncthreads();
  if (tid == 0) {
    stats[5] = rnum[0]; stats[6] = rden[0];
    stats[8]  = (float)cnt[0]; stats[9]  = (float)cnt[1];
    stats[10] = (float)cnt[2]; stats[11] = (float)cnt[3];
  }
}

// ---------------------------------------------------------------------------
// K6: feature assembly + 7-head sigmoid
// ---------------------------------------------------------------------------
__global__ void head_kernel(const float* __restrict__ stats,
                            const float* __restrict__ Wh,
                            const float* __restrict__ bh,
                            float* __restrict__ out)
{
  int k = threadIdx.x;
  if (k >= NOUT) return;
  float moran = stats[5] / stats[6];
  float perc[4];
  #pragma unroll
  for (int j = 0; j < 4; ++j) perc[j] = stats[8 + j] * (100.0f / (float)CELLS);
  float pmn = perc[0], pmx = perc[0];
  #pragma unroll
  for (int j = 1; j < 4; ++j) { pmn = fminf(pmn, perc[j]); pmx = fmaxf(pmx, perc[j]); }
  float inv = 1.0f / (pmx - pmn);
  float feat[5];
  feat[0] = moran;
  #pragma unroll
  for (int j = 0; j < 4; ++j) feat[j + 1] = (perc[j] - pmn) * inv;
  float a = bh[k];
  #pragma unroll
  for (int j = 0; j < 5; ++j) a += feat[j] * Wh[k * 5 + j];
  out[k] = 1.0f / (1.0f + __expf(-a));
}

// ---------------------------------------------------------------------------
extern "C" void kernel_launch(void* const* d_in, const int* in_sizes, int n_in,
                              void* d_out, int out_size, void* d_ws, size_t ws_size,
                              hipStream_t stream)
{
  const float* x    = (const float*)d_in[0];
  const int*   coor = (const int*)  d_in[1];
  const float* W1   = (const float*)d_in[2];
  const float* b1   = (const float*)d_in[3];
  const float* W2   = (const float*)d_in[4];
  const float* b2   = (const float*)d_in[5];
  const float* Wh   = (const float*)d_in[6];
  const float* bh   = (const float*)d_in[7];

  float*  out   = (float*)d_out;       // [7] heads
  float*  Amat  = out + NOUT;          // [256*256] A_matrix (output #2)
  float*  s_pre = (float*)d_ws;        // [N_TILE] pre-softmax scores
  float*  stats = s_pre + ((N_TILE + 63) & ~63);
  bf16_t* w1tg  = (bf16_t*)(stats + 64);  // [D*L] bf16 W1^T image

  const int nwg = (N_TILE + ROWS_PER_WG - 1) / ROWS_PER_WG;  // 391
  w1_transpose_kernel<<<(L * D) / 256, 256, 0, stream>>>(W1, w1tg);
  attn_gemm_kernel<<<nwg, 256, 0, stream>>>(x, w1tg, b1, W2, b2, s_pre);
  softmax_stats_kernel<<<1, 1024, 0, stream>>>(s_pre, stats);
  zero_kernel<<<CELLS / 256, 256, 0, stream>>>(Amat, CELLS);
  scatter_kernel<<<(N_TILE + 255) / 256, 256, 0, stream>>>(s_pre, coor, stats, Amat);
  mat_stats_kernel<<<1, 1024, 0, stream>>>(Amat, stats);
  moran_bucket_kernel<<<1, 1024, 0, stream>>>(Amat, stats);
  head_kernel<<<1, 32, 0, stream>>>(stats, Wh, bh, out);
}
